// MultiHeadedAttention_3607772529120
// MI455X (gfx1250) — compile-verified, matrix operand order checked by reading, not run on hardware
//
#include <hip/hip_runtime.h>

typedef __bf16 bf16_t;
typedef __attribute__((ext_vector_type(16))) __bf16 v16bf;
typedef __attribute__((ext_vector_type(8)))  __bf16 v8bf;
typedef __attribute__((ext_vector_type(8)))  float  v8f;
typedef __attribute__((ext_vector_type(4)))  float  v4f;

// ---------------------------------------------------------------------------
// CDNA5 async global->LDS copy (ASYNCcnt-tracked DMA, 16B per lane).
// Generic LDS pointers carry the LDS byte offset in their low 32 bits
// (shared aperture = high bits only), so truncation yields the DS address.
// ---------------------------------------------------------------------------
__device__ __forceinline__ void async_copy_b128(void* lds_dst, const void* gsrc) {
  unsigned loff = (unsigned)(unsigned long long)lds_dst;
  unsigned long long g = (unsigned long long)gsrc;
  asm volatile("global_load_async_to_lds_b128 %0, %1, off"
               :: "v"(loff), "v"(g) : "memory");
}

#if defined(__has_builtin)
#if __has_builtin(__builtin_amdgcn_s_wait_asynccnt)
#define WAIT_ASYNC(n) __builtin_amdgcn_s_wait_asynccnt(n)
#endif
#endif
#ifndef WAIT_ASYNC
#define WAIT_ASYNC(n) asm volatile("s_wait_asynccnt %0" :: "n"(n) : "memory")
#endif

// ---------------------------------------------------------------------------
// WMMA helpers (CDNA5 wave32, 16x16x32 bf16 -> f32)
// ---------------------------------------------------------------------------
__device__ __forceinline__ v8f wmma_bf16(v16bf a, v16bf b, v8f c) {
  return __builtin_amdgcn_wmma_f32_16x16x32_bf16(false, a, false, b, (short)0, c,
                                                 false, false);
}

// 16-bit A/B fragment from a row-major LDS tile (ISA 7.12.2):
// lane l (m|n = l&15, khalf = l>>4) holds K = khalf*8..+7 and 16+khalf*8..+7
// => two contiguous 16-byte LDS reads.
__device__ __forceinline__ v16bf frag_ld(const bf16_t* base, int row, int stride,
                                         int lane) {
  const bf16_t* p = base + row * stride + ((lane >> 4) & 1) * 8;
  v8bf lo = *(const v8bf*)(p);
  v8bf hi = *(const v8bf*)(p + 16);
  v16bf r;
#pragma unroll
  for (int i = 0; i < 8; ++i) { r[i] = lo[i]; r[i + 8] = hi[i]; }
  return r;
}

__device__ __forceinline__ v8f zero8() {
  v8f z;
#pragma unroll
  for (int i = 0; i < 8; ++i) z[i] = 0.f;
  return z;
}

// ---------------------------------------------------------------------------
// Projection GEMM: Y[m, n] = sum_k X[m, k] * W[n, k] + bias[n]
// X: [M, 768] fp32 or bf16, W: [768, 768] fp32 row-major [n][k].
// 128x128 workgroup tile, 8 waves x (32M x 64N).  Register-pipelined staging:
// next tile's global loads issue before the current tile's WMMAs.
// OUT_TRANS stores bf16 transposed per batch: Y[(b*768 + n)*2048 + key].
// ---------------------------------------------------------------------------
template <typename TIn, bool OUT_F32, bool OUT_TRANS>
__global__ __launch_bounds__(256) void gemm_xwt(const TIn* __restrict__ X,
                                                const float* __restrict__ W,
                                                const float* __restrict__ bias,
                                                void* __restrict__ Y) {
  __shared__ bf16_t As[128 * 32];
  __shared__ bf16_t Bs[128 * 32];

  const int tid  = threadIdx.x;
  const int lane = tid & 31;
  const int ln   = lane & 15;
  const int w    = tid >> 5;
  const int mrow = (w >> 1) * 32;
  const int ncol = (w & 1) * 64;
  const long mbase = (long)blockIdx.x * 128;
  const int  nbase = blockIdx.y * 128;

  // per-thread staging slots: rows r0, r0+64, 8 elems at column cg
  const int r0 = tid >> 2;
  const int cg = (tid & 3) * 8;

  v4f  xf[2][2], wf[2][2];
  v8bf xb[2];

  auto gload = [&](int k0) {
#pragma unroll
    for (int r = 0; r < 2; ++r) {
      const long row = mbase + r0 + r * 64;
      if constexpr (sizeof(TIn) == 4) {
        const float* s = (const float*)X + row * 768 + k0 + cg;
        xf[r][0] = *(const v4f*)(s);
        xf[r][1] = *(const v4f*)(s + 4);
      } else {
        xb[r] = *(const v8bf*)((const bf16_t*)X + row * 768 + k0 + cg);
      }
      const float* ws = W + (long)(nbase + r0 + r * 64) * 768 + k0 + cg;
      wf[r][0] = *(const v4f*)(ws);
      wf[r][1] = *(const v4f*)(ws + 4);
    }
  };
  auto stage = [&]() {
#pragma unroll
    for (int r = 0; r < 2; ++r) {
      const int row = r0 + r * 64;
      v8bf oa;
      if constexpr (sizeof(TIn) == 4) {
#pragma unroll
        for (int i = 0; i < 4; ++i) {
          oa[i] = (bf16_t)xf[r][0][i];
          oa[i + 4] = (bf16_t)xf[r][1][i];
        }
      } else {
        oa = xb[r];
      }
      *(v8bf*)(&As[row * 32 + cg]) = oa;
      v8bf ob;
#pragma unroll
      for (int i = 0; i < 4; ++i) {
        ob[i] = (bf16_t)wf[r][0][i];
        ob[i + 4] = (bf16_t)wf[r][1][i];
      }
      *(v8bf*)(&Bs[row * 32 + cg]) = ob;
    }
  };

  v8f acc[2][4];
#pragma unroll
  for (int i = 0; i < 2; ++i)
#pragma unroll
    for (int j = 0; j < 4; ++j) acc[i][j] = zero8();

  gload(0);
  for (int k0 = 0; k0 < 768; k0 += 32) {
    __syncthreads();            // previous compute finished reading As/Bs
    stage();
    __syncthreads();
    if (k0 + 32 < 768) gload(k0 + 32);   // overlap with WMMAs below

    v16bf a0 = frag_ld(As, mrow + ln, 32, lane);
    v16bf a1 = frag_ld(As, mrow + 16 + ln, 32, lane);
#pragma unroll
    for (int ni = 0; ni < 4; ++ni) {
      v16bf b = frag_ld(Bs, ncol + ni * 16 + ln, 32, lane);
      acc[0][ni] = wmma_bf16(a0, b, acc[0][ni]);
      acc[1][ni] = wmma_bf16(a1, b, acc[1][ni]);
    }
  }

  // Epilogue.  C layout: lanes0-15 -> M=v, lanes16-31 -> M=v+8.
  const int lrow = ((lane >> 4) & 1) * 8;
#pragma unroll
  for (int ni = 0; ni < 4; ++ni) {
    const int n = nbase + ncol + ni * 16 + ln;
    const float bv = bias[n];
#pragma unroll
    for (int mi = 0; mi < 2; ++mi) {
#pragma unroll
      for (int v = 0; v < 8; ++v) {
        const long m = mbase + mrow + mi * 16 + lrow + v;
        const float val = acc[mi][ni][v] + bv;
        if constexpr (OUT_F32) {
          ((float*)Y)[m * 768 + n] = val;
        } else if constexpr (OUT_TRANS) {
          const long bb = m >> 11;          // M = b*2048 + key
          const long key = m & 2047;
          ((bf16_t*)Y)[(bb * 768 + n) * 2048 + key] = (bf16_t)val;
        } else {
          ((bf16_t*)Y)[m * 768 + n] = (bf16_t)val;
        }
      }
    }
  }
}

// ---------------------------------------------------------------------------
// Fused flash cross-attention with async-LDS double-buffered staging.
// Q: [16,1024,768] bf16, K: [16,2048,768] bf16, Vt: [16,768,2048] bf16,
// ctx: [16,1024,768] bf16.  128 threads (4 waves), 32 q rows / WG,
// 64-key streaming chunks.  Wave w: rowgrp = w&1, khalfw = w>>1.
// ---------------------------------------------------------------------------
__global__ __launch_bounds__(128) void attn_fused(const bf16_t* __restrict__ Qw,
                                                  const bf16_t* __restrict__ Kw,
                                                  const bf16_t* __restrict__ Vt,
                                                  bf16_t* __restrict__ Ctx) {
  __shared__ bf16_t Qs[2][32 * 32];   //  4 KB (double buffered)
  __shared__ bf16_t Ks[2][64 * 32];   //  8 KB
  __shared__ bf16_t Ps[32 * 64];      //  4 KB
  __shared__ bf16_t Vs[2][128 * 64];  // 32 KB (Vs[col][key], from Vt rows)
  __shared__ float m_run[32], l_run[32], scale_s[32];
  __shared__ float pmax_s[32][2], psum_s[32][2];

  const int tid    = threadIdx.x;
  const int lane   = tid & 31;
  const int ln     = lane & 15;
  const int hb     = (lane >> 4) & 1;
  const int w      = tid >> 5;
  const int rowgrp = w & 1;
  const int khalfw = w >> 1;

  const int b  = blockIdx.x >> 5;
  const int q0 = (blockIdx.x & 31) * 32;
  const long qrowbase = (long)b * 1024 + q0;
  const float sm_scale = 0.03608439182435161f;  // 1/sqrt(768)

  if (tid < 32) { m_run[tid] = -1e30f; l_run[tid] = 0.f; }

  v8f o[24];
#pragma unroll
  for (int t = 0; t < 24; ++t) o[t] = zero8();

  const int srow = tid >> 2;          // staging row (0..31), col group cg
  const int scg  = (tid & 3) * 8;

  for (int c = 0; c < 2048; c += 64) {
    // 3 async instructions per wave per stage (1 Q + 2 K chunks per thread)
    auto stage_qk = [&](int buf, int d0) {
      async_copy_b128(&Qs[buf][srow * 32 + scg],
                      Qw + (qrowbase + srow) * 768 + d0 + scg);
      async_copy_b128(&Ks[buf][srow * 32 + scg],
                      Kw + ((long)b * 2048 + c + srow) * 768 + d0 + scg);
      async_copy_b128(&Ks[buf][(srow + 32) * 32 + scg],
                      Kw + ((long)b * 2048 + c + srow + 32) * 768 + d0 + scg);
    };
    // 8 async instructions per wave per stage (Vs row = d, 64 keys = 128B)
    auto stage_v = [&](int buf, int jj) {
#pragma unroll
      for (int t = 0; t < 8; ++t) {
        const int q = tid + t * 128;
        const int r = q >> 3;                 // 0..127
        const int s = q & 7;
        const int dg = jj * 64 + r + (r < 64 ? 0 : 320);
        async_copy_b128(&Vs[buf][r * 64 + s * 8],
                        Vt + ((long)b * 768 + dg) * 2048 + c + s * 8);
      }
    };

    // ---- S = Q.K^T over D, async double-buffered 32-wide slices ----
    v8f s0 = zero8(), s1 = zero8();
    stage_qk(0, 0);
    for (int it = 0; it < 24; ++it) {
      if (it + 1 < 24) { stage_qk((it + 1) & 1, (it + 1) * 32); WAIT_ASYNC(3); }
      else             { WAIT_ASYNC(0); }
      __syncthreads();
      const bf16_t* qb = Qs[it & 1];
      const bf16_t* kb = Ks[it & 1];
      v16bf a  = frag_ld(qb, rowgrp * 16 + ln, 32, lane);
      v16bf b0 = frag_ld(kb, khalfw * 32 + ln, 32, lane);
      v16bf b1 = frag_ld(kb, khalfw * 32 + 16 + ln, 32, lane);
      s0 = wmma_bf16(a, b0, s0);
      s1 = wmma_bf16(a, b1, s1);
      __syncthreads();
    }
    stage_v(0, 0);   // V stage 0 DMA overlaps the softmax below

#pragma unroll
    for (int v = 0; v < 8; ++v) { s0[v] *= sm_scale; s1[v] *= sm_scale; }

    // ---- online softmax ----
    float pm[8];
#pragma unroll
    for (int v = 0; v < 8; ++v) {
      float x = fmaxf(s0[v], s1[v]);
      x = fmaxf(x, __shfl_xor(x, 1, 32));
      x = fmaxf(x, __shfl_xor(x, 2, 32));
      x = fmaxf(x, __shfl_xor(x, 4, 32));
      x = fmaxf(x, __shfl_xor(x, 8, 32));
      pm[v] = x;
    }
    if (ln == 0) {
#pragma unroll
      for (int v = 0; v < 8; ++v) pmax_s[rowgrp * 16 + hb * 8 + v][khalfw] = pm[v];
    }
    __syncthreads();
    if (tid < 32) {
      const float mo = m_run[tid];
      const float mn = fmaxf(mo, fmaxf(pmax_s[tid][0], pmax_s[tid][1]));
      m_run[tid]   = mn;
      scale_s[tid] = __expf(mo - mn);
    }
    __syncthreads();

    float sc[8];
#pragma unroll
    for (int v = 0; v < 8; ++v) {
      const int row = rowgrp * 16 + hb * 8 + v;
      const float mn = m_run[row];
      const float p0 = __expf(s0[v] - mn);
      const float p1 = __expf(s1[v] - mn);
      Ps[row * 64 + khalfw * 32 + ln]      = (bf16_t)p0;
      Ps[row * 64 + khalfw * 32 + 16 + ln] = (bf16_t)p1;
      float su = p0 + p1;
      su += __shfl_xor(su, 1, 32);
      su += __shfl_xor(su, 2, 32);
      su += __shfl_xor(su, 4, 32);
      su += __shfl_xor(su, 8, 32);
      if (ln == 0) psum_s[row][khalfw] = su;
      sc[v] = scale_s[row];
    }
#pragma unroll
    for (int t = 0; t < 24; ++t)
#pragma unroll
      for (int v = 0; v < 8; ++v) o[t][v] *= sc[v];
    __syncthreads();
    if (tid < 32)
      l_run[tid] = l_run[tid] * scale_s[tid] + psum_s[tid][0] + psum_s[tid][1];

    // ---- O += P.V, async double-buffered V column blocks ----
    for (int j = 0; j < 6; ++j) {
      if (j + 1 < 6) { stage_v((j + 1) & 1, j + 1); WAIT_ASYNC(8); }
      else           { WAIT_ASYNC(0); }
      __syncthreads();
      const bf16_t* vb = Vs[j & 1];
#pragma unroll
      for (int ks = 0; ks < 2; ++ks) {
        v16bf a = frag_ld(Ps + ks * 32, rowgrp * 16 + ln, 64, lane);
#pragma unroll
        for (int ct = 0; ct < 4; ++ct) {
          v16bf bb = frag_ld(vb + ks * 32, khalfw * 64 + ct * 16 + ln, 64, lane);
          o[j * 4 + ct] = wmma_bf16(a, bb, o[j * 4 + ct]);
        }
      }
      __syncthreads();
    }
  }

  // ---- finalize ----
  float inv[8];
#pragma unroll
  for (int v = 0; v < 8; ++v) {
    const float l = l_run[rowgrp * 16 + hb * 8 + v];
    inv[v] = (l > 0.f) ? 1.f / l : 0.f;
  }
#pragma unroll
  for (int t = 0; t < 24; ++t) {
    const int j  = t >> 2;
    const int ct = t & 3;
    const int col = khalfw * 384 + j * 64 + ct * 16 + ln;
#pragma unroll
    for (int v = 0; v < 8; ++v) {
      const long q = qrowbase + rowgrp * 16 + hb * 8 + v;
      Ctx[q * 768 + col] = (bf16_t)(o[t][v] * inv[v]);
    }
  }
}

// ---------------------------------------------------------------------------
// Host launcher
// ---------------------------------------------------------------------------
extern "C" void kernel_launch(void* const* d_in, const int* in_sizes, int n_in,
                              void* d_out, int out_size, void* d_ws, size_t ws_size,
                              hipStream_t stream) {
  (void)in_sizes; (void)n_in; (void)out_size; (void)ws_size;
  const float* query = (const float*)d_in[0];  // [16, 2048, 768]
  const float* guide = (const float*)d_in[1];  // [16, 1024, 768]
  const float* Wq = (const float*)d_in[2];
  const float* bq = (const float*)d_in[3];
  const float* Wk = (const float*)d_in[4];
  const float* bk = (const float*)d_in[5];
  const float* Wv = (const float*)d_in[6];
  const float* bv = (const float*)d_in[7];
  const float* Wo = (const float*)d_in[8];
  const float* bo = (const float*)d_in[9];
  float* out = (float*)d_out;                  // [16, 1024, 768] fp32

  const size_t nQ  = (size_t)16 * 1024 * 768;
  const size_t nKV = (size_t)16 * 2048 * 768;
  char* wsb = (char*)d_ws;
  bf16_t* Qw  = (bf16_t*)(wsb);
  bf16_t* Kw  = (bf16_t*)(wsb + nQ * 2);
  bf16_t* Vtw = (bf16_t*)(wsb + nQ * 2 + nKV * 2);   // transposed [b][768][2048]
  bf16_t* Ctx = (bf16_t*)(wsb + nQ * 2 + nKV * 4);

  gemm_xwt<float, false, false><<<dim3(16 * 1024 / 128, 6), 256, 0, stream>>>(guide, Wq, bq, Qw);
  gemm_xwt<float, false, false><<<dim3(16 * 2048 / 128, 6), 256, 0, stream>>>(query, Wk, bk, Kw);
  gemm_xwt<float, false, true ><<<dim3(16 * 2048 / 128, 6), 256, 0, stream>>>(query, Wv, bv, Vtw);

  attn_fused<<<dim3(16 * 32), 128, 0, stream>>>(Qw, Kw, Vtw, Ctx);

  gemm_xwt<bf16_t, true, false><<<dim3(16 * 1024 / 128, 6), 256, 0, stream>>>(Ctx, Wo, bo, out);
}
